// LightFactorFusion_87385404604945
// MI455X (gfx1250) — matrix operand order repaired and checked by reading, not compiled, not run on hardware
//
#include <hip/hip_runtime.h>
#include <stdint.h>

typedef __attribute__((ext_vector_type(16))) __bf16 v16bf;
typedef __attribute__((ext_vector_type(8)))  float  v8f;

#define DD     512
#define MT     16
#define RPAD   64
#define V8Z    {0.f,0.f,0.f,0.f,0.f,0.f,0.f,0.f}

__device__ __forceinline__ unsigned short f2bf(float f) {
  unsigned u = __float_as_uint(f);
  unsigned r = u + 0x7FFFu + ((u >> 16) & 1u);   // round-to-nearest-even
  return (unsigned short)(r >> 16);
}

union ABFrag { v16bf v; unsigned u[8]; };

// A fragment (contiguous K order, matching B): lanes 0-15 -> M=lane, K {0..15};
// lanes 16-31 -> M=lane-16, K {16..31}. LDS rows are contiguous ushort.
__device__ __forceinline__ v16bf load_a_frag(const unsigned short* lds, int rowStride,
                                             int kbase, int lane) {
  int m   = lane & 15;
  int off = (lane & 16) ? 16 : 0;
  const unsigned* p = (const unsigned*)(lds + m * rowStride + kbase);
  ABFrag f;
  int b0 = off >> 1;
#pragma unroll
  for (int i = 0; i < 8; ++i) { f.u[i] = p[b0 + i]; }
  return f.v;
}

// B fragments are pre-swizzled by prep_kernel: per (tile,lane) 16 contiguous bf16.
__device__ __forceinline__ v16bf load_b_frag(const unsigned short* gfrag, int tile, int lane) {
  const uint4* p = (const uint4*)gfrag + ((size_t)tile * 32 + lane) * 2;
  uint4 a = p[0], b = p[1];
  ABFrag f;
  f.u[0]=a.x; f.u[1]=a.y; f.u[2]=a.z; f.u[3]=a.w;
  f.u[4]=b.x; f.u[5]=b.y; f.u[6]=b.z; f.u[7]=b.w;
  return f.v;
}

__device__ __forceinline__ unsigned fkey(float s) {
  unsigned b = __float_as_uint(s);
  return b ^ (unsigned)(((int)b >> 31) | 0x80000000);
}

// ---- Prep: bf16-convert + swizzle W_sel^T, U(pad 64), V(pad 64) into B-fragment order ----
__global__ void __launch_bounds__(256) prep_kernel(const float* __restrict__ Wsel,
    const float* __restrict__ U, const float* __restrict__ V,
    unsigned short* __restrict__ wselB, unsigned short* __restrict__ uB,
    unsigned short* __restrict__ vB) {
  int e = blockIdx.x * 256 + threadIdx.x;
  if (e < 262144) {                       // W_sel: 32 nt x 16 ks tiles, B[k][n]=W_sel[n][k]
    int tile = e >> 9, w = e & 511;
    int lane = w >> 4, q = w & 15;
    int nt = tile >> 4, ks = tile & 15;
    int n = nt * 16 + (lane & 15);
    int k = ks * 32 + ((lane & 16) ? 16 : 0) + q;
    wselB[e] = f2bf(Wsel[n * DD + k]);
  } else if (e < 262144 + 32768) {        // U: 4 nt x 16 ks, B[k][r]=U[k][r], pad r>=51
    int e2 = e - 262144;
    int tile = e2 >> 9, w = e2 & 511;
    int lane = w >> 4, q = w & 15;
    int nt = tile >> 4, ks = tile & 15;
    int r = nt * 16 + (lane & 15);
    int k = ks * 32 + ((lane & 16) ? 16 : 0) + q;
    uB[e2] = (r < 51) ? f2bf(U[k * 51 + r]) : (unsigned short)0;
  } else if (e < 262144 + 65536) {        // V: 32 nt x 2 ks, B[k][n]=V[k][n], pad k>=51
    int e2 = e - 262144 - 32768;
    int tile = e2 >> 9, w = e2 & 511;
    int lane = w >> 4, q = w & 15;
    int nt = tile >> 1, ks = tile & 1;
    int n = nt * 16 + (lane & 15);
    int k = ks * 32 + ((lane & 16) ? 16 : 0) + q;
    vB[e2] = (k < 51) ? f2bf(V[k * DD + n]) : (unsigned short)0;
  }
}

// ---- Fused: scores GEMM -> top-k mask -> low-rank cross -> gated residual ----
__global__ void __launch_bounds__(256) fused_kernel(const float* __restrict__ x,
    const float* __restrict__ bsel, const float* __restrict__ Wg,
    const float* __restrict__ bg, const int* __restrict__ kptr,
    const unsigned short* __restrict__ wselB, const unsigned short* __restrict__ uB,
    const unsigned short* __restrict__ vB, float* __restrict__ out) {
  extern __shared__ char smem[];
  float*          xs  = (float*)smem;                          // 16x512 f32: x -> x_sparse
  float*          sc  = (float*)(smem + 32768);                // 16x512 f32: scores -> x_inter
  unsigned short* xbh = (unsigned short*)(smem + 65536);       // 16x512 bf16
  unsigned short* tbh = (unsigned short*)(smem + 65536 + 16384); // 16x64 bf16 (t = xs@U)
  unsigned*       thr = (unsigned*)(smem + 65536 + 16384 + 2048); // 16 thresholds

  const int t    = threadIdx.x;
  const int lane = t & 31;
  const int wid  = t >> 5;
  const int row0 = blockIdx.x * MT;
  const int kk   = kptr[0];

  // Stage 0: async-copy x tile (16x512 f32 = 32KB) into LDS.
  {
    const char* gx = (const char*)(x + (size_t)row0 * DD);
#pragma unroll
    for (int i = 0; i < 8; ++i) {
      unsigned idx = t + i * 256;                               // uint4 index, 2048 total
      unsigned ldsOff = (unsigned)(uintptr_t)(void*)(smem + (size_t)idx * 16);
      unsigned long long ga = (unsigned long long)(uintptr_t)gx + (unsigned long long)idx * 16ull;
      asm volatile("global_load_async_to_lds_b128 %0, %1, off"
                   :: "v"(ldsOff), "v"(ga) : "memory");
    }
    asm volatile("s_wait_asynccnt 0" ::: "memory");
  }
  __syncthreads();

  // bf16 copy of x for WMMA A operand
#pragma unroll
  for (int i = 0; i < 32; ++i) { int idx = t + i * 256; xbh[idx] = f2bf(xs[idx]); }
  __syncthreads();

  // Stage 1: logits = x @ W_sel^T + b_sel (sigmoid dropped: monotonic, top-k invariant)
  {
    v8f acc[4] = {V8Z, V8Z, V8Z, V8Z};
    for (int ks = 0; ks < 16; ++ks) {
      v16bf a = load_a_frag(xbh, DD, ks * 32, lane);
      if (ks + 1 < 16)
        __builtin_prefetch(wselB + (((size_t)(wid * 4) * 16 + ks + 1) * 32 + lane) * 16, 0, 1);
#pragma unroll
      for (int i = 0; i < 4; ++i) {
        int nt = wid * 4 + i;
        v16bf b = load_b_frag(wselB, nt * 16 + ks, lane);
        acc[i] = __builtin_amdgcn_wmma_f32_16x16x32_bf16(
            false, a, false, b, (short)0, acc[i], false, false);
      }
    }
    int mbase = (lane & 16) ? 8 : 0;
#pragma unroll
    for (int i = 0; i < 4; ++i) {
      int n = (wid * 4 + i) * 16 + (lane & 15);
      float bias = bsel[n];
#pragma unroll
      for (int r = 0; r < 8; ++r) sc[(mbase + r) * DD + n] = acc[i][r] + bias;
    }
  }
  __syncthreads();

  // Stage 2: exact top-k threshold per row (radix select on order-preserving keys)
#pragma unroll
  for (int rr = 0; rr < 2; ++rr) {
    int row = wid * 2 + rr;
    unsigned keys[16];
#pragma unroll
    for (int i = 0; i < 16; ++i) keys[i] = fkey(sc[row * DD + lane + i * 32]);
    unsigned T = 0;
    for (int bit = 31; bit >= 0; --bit) {
      unsigned cand = T | (1u << bit);
      int c = 0;
#pragma unroll
      for (int i = 0; i < 16; ++i) c += (keys[i] >= cand) ? 1 : 0;
#pragma unroll
      for (int off = 16; off >= 1; off >>= 1) c += __shfl_xor(c, off, 32);
      if (c >= kk) T = cand;
    }
    if (lane == 0) thr[row] = T;
  }
  __syncthreads();

  // Stage 3: mask -> x_sparse (fp32 + bf16)
#pragma unroll
  for (int i = 0; i < 32; ++i) {
    int idx = t + i * 256;
    int m = idx >> 9;
    float xv = (fkey(sc[idx]) >= thr[m]) ? xs[idx] : 0.0f;
    xs[idx]  = xv;
    xbh[idx] = f2bf(xv);
  }
  __syncthreads();

  // Stage 4a: t = x_sparse @ U  (16x512x64, waves 0..3 each own one N-tile)
  if (wid < 4) {
    v8f acc = V8Z;
    for (int ks = 0; ks < 16; ++ks) {
      v16bf a = load_a_frag(xbh, DD, ks * 32, lane);
      v16bf b = load_b_frag(uB, wid * 16 + ks, lane);
      acc = __builtin_amdgcn_wmma_f32_16x16x32_bf16(
          false, a, false, b, (short)0, acc, false, false);
    }
    int n = wid * 16 + (lane & 15);
    int mbase = (lane & 16) ? 8 : 0;
#pragma unroll
    for (int r = 0; r < 8; ++r) tbh[(mbase + r) * RPAD + n] = f2bf(acc[r]);
  }
  __syncthreads();

  // Stage 4b: cross = t @ V, fused x_inter = x_sparse * (1 + scale*cross)
  {
    const float scale = 0.14002800840280098f;  // 1/sqrt(51)
#pragma unroll
    for (int i = 0; i < 4; ++i) {
      int nt = wid * 4 + i;
      v8f acc = V8Z;
#pragma unroll
      for (int ks = 0; ks < 2; ++ks) {
        v16bf a = load_a_frag(tbh, RPAD, ks * 32, lane);
        v16bf b = load_b_frag(vB, nt * 2 + ks, lane);
        acc = __builtin_amdgcn_wmma_f32_16x16x32_bf16(
            false, a, false, b, (short)0, acc, false, false);
      }
      int n = nt * 16 + (lane & 15);
      int mbase = (lane & 16) ? 8 : 0;
#pragma unroll
      for (int r = 0; r < 8; ++r) {
        int m = mbase + r;
        float xv = xs[m * DD + n];
        sc[m * DD + n] = xv * (1.0f + scale * acc[r]);
      }
    }
  }
  __syncthreads();

  // Stage 5: gate = sigmoid(x_inter . W_gate + b_gate); out = g*x_inter + (1-g)*x_sparse
  {
    float bgv = bg[0];
#pragma unroll
    for (int rr = 0; rr < 2; ++rr) {
      int row = wid * 2 + rr;
      float dot = 0.f;
#pragma unroll
      for (int i = 0; i < 16; ++i) { int d = lane + i * 32; dot += sc[row * DD + d] * Wg[d]; }
#pragma unroll
      for (int off = 16; off >= 1; off >>= 1) dot += __shfl_xor(dot, off, 32);
      float g = 1.0f / (1.0f + __expf(-(dot + bgv)));
      float* orow = out + (size_t)(row0 + row) * DD;
#pragma unroll
      for (int i = 0; i < 16; ++i) {
        int d = lane + i * 32;
        orow[d] = g * sc[row * DD + d] + (1.0f - g) * xs[row * DD + d];
      }
    }
  }
}

extern "C" void kernel_launch(void* const* d_in, const int* in_sizes, int n_in,
                              void* d_out, int out_size, void* d_ws, size_t ws_size,
                              hipStream_t stream) {
  (void)n_in; (void)out_size; (void)ws_size;
  const float* x    = (const float*)d_in[0];
  const float* Wsel = (const float*)d_in[1];
  const float* bsel = (const float*)d_in[2];
  const float* U    = (const float*)d_in[3];
  const float* V    = (const float*)d_in[4];
  const float* Wg   = (const float*)d_in[5];
  const float* bg   = (const float*)d_in[6];
  const int*   kptr = (const int*)d_in[7];
  float* out = (float*)d_out;

  unsigned short* wselB = (unsigned short*)d_ws;     // 512 KB
  unsigned short* uB    = wselB + 262144;            //  64 KB
  unsigned short* vB    = uB + 32768;                //  64 KB

  prep_kernel<<<1280, 256, 0, stream>>>(Wsel, U, V, wselB, uB, vB);

  int Brows  = in_sizes[0] / DD;
  int blocks = Brows / MT;
  size_t lds = 32768 + 32768 + 16384 + 2048 + 64;    // ~84 KB
  fused_kernel<<<blocks, 256, lds, stream>>>(x, bsel, Wg, bg, kptr, wselB, uB, vB, out);
}
